// MoETransformer_53730040873135
// MI455X (gfx1250) — matrix operand order repaired and checked by reading, not compiled, not run on hardware
//
#include <hip/hip_runtime.h>
#include <hip/hip_bf16.h>
#include <math.h>

// ---- problem constants (match reference) ----
#define B_   2
#define T_   1024
#define C_   512
#define NH_  8
#define HD_  64
#define L_   4
#define E_   8
#define K_   2
#define V_   50257
#define FF_  2048
#define NTOK (B_*T_)
#define NASSIGN (NTOK*K_)

typedef __attribute__((ext_vector_type(16))) __bf16 v16bf;
typedef __attribute__((ext_vector_type(8)))  __bf16 v8bf;
typedef __attribute__((ext_vector_type(8)))  float  v8f;
typedef __attribute__((ext_vector_type(4)))  float  v4f;

#define FRAG_SZ 512   // bf16 elements per 16x32 (A) / 32x16 (B) fragment: 32 lanes * 16

// f32 -> bf16 via native convert (RNE); backend should emit v_cvt_*bf16* ops
static __device__ __forceinline__ __bf16 f2bf(float f){
  return (__bf16)f;
}

static __device__ __forceinline__ v8f wmma_bf16(v16bf a, v16bf b, v8f c){
  return __builtin_amdgcn_wmma_f32_16x16x32_bf16(false, a, false, b, (short)0, c, false, false);
}

// Fragment-ready LDS: each lane's 16 bf16 elements are contiguous (32B) -> ds_load_b128 x2
static __device__ __forceinline__ v16bf load_frag(const __bf16* fragbase){
  return *(const v16bf*)(fragbase + (threadIdx.x & 31)*16);
}

static __device__ __forceinline__ float gelu_new(float x){
  float x3 = x*x*x;
  return 0.5f*x*(1.0f + tanhf(0.7978845608028654f*(x + 0.044715f*x3)));
}

// ---------------- elementwise / small kernels ----------------

__global__ void embed_kernel(const int* __restrict__ idx, const float* __restrict__ wte,
                             const float* __restrict__ wpe, float* __restrict__ x){
  int i = blockIdx.x*256 + threadIdx.x;           // i < NTOK*C_
  int tok = i / C_, c = i - tok*C_;
  int t = tok % T_;
  x[i] = wte[(size_t)idx[tok]*C_ + c] + wpe[(size_t)t*C_ + c];
}

__global__ void add_kernel(float* __restrict__ x, const float* __restrict__ y, int n){
  int i = blockIdx.x*256 + threadIdx.x;
  if (i < n) x[i] += y[i];
}

__global__ void ln_kernel(const float* __restrict__ xin, const float* __restrict__ g,
                          const float* __restrict__ bta, float* __restrict__ out){
  const int row = blockIdx.x, tid = threadIdx.x;
  __shared__ float sb[256];
  const float* xr = xin + (size_t)row*C_;
  float s = 0.f;
  for (int c = tid; c < C_; c += 256) s += xr[c];
  sb[tid] = s; __syncthreads();
  for (int o = 128; o; o >>= 1){ if (tid < o) sb[tid] += sb[tid+o]; __syncthreads(); }
  float mu = sb[0] * (1.0f/C_);
  __syncthreads();
  float v = 0.f;
  for (int c = tid; c < C_; c += 256){ float d = xr[c]-mu; v += d*d; }
  sb[tid] = v; __syncthreads();
  for (int o = 128; o; o >>= 1){ if (tid < o) sb[tid] += sb[tid+o]; __syncthreads(); }
  float inv = rsqrtf(sb[0]*(1.0f/C_) + 1e-5f);
  for (int c = tid; c < C_; c += 256)
    out[(size_t)row*C_ + c] = (xr[c]-mu)*inv*g[c] + bta[c];
}

__global__ void zero_counts_kernel(int* counts){
  if (threadIdx.x < E_) counts[threadIdx.x] = 0;
}

__global__ void router_kernel(const float* __restrict__ xn, const float* __restrict__ wr,
                              int* __restrict__ topi, float* __restrict__ gatev,
                              int* __restrict__ counts){
  const int token = blockIdx.x*8 + (threadIdx.x >> 5);
  const int lane  = threadIdx.x & 31;
  if (token >= NTOK) return;
  float p[E_];
#pragma unroll
  for (int e = 0; e < E_; ++e) p[e] = 0.f;
  const float* xr = xn + (size_t)token*C_;
  for (int c = lane; c < C_; c += 32){
    float xv = xr[c];
    const float* w = &wr[(size_t)c*E_];
#pragma unroll
    for (int e = 0; e < E_; ++e) p[e] += xv * w[e];
  }
#pragma unroll
  for (int off = 16; off; off >>= 1)
#pragma unroll
    for (int e = 0; e < E_; ++e) p[e] += __shfl_xor(p[e], off, 32);
  if (lane == 0){
    float mx = p[0];
#pragma unroll
    for (int e = 1; e < E_; ++e) mx = fmaxf(mx, p[e]);
    float q[E_], s = 0.f;
#pragma unroll
    for (int e = 0; e < E_; ++e){ q[e] = __expf(p[e]-mx); s += q[e]; }
    float inv = 1.0f/s;
#pragma unroll
    for (int e = 0; e < E_; ++e) q[e] *= inv;
    int i0 = 0;
#pragma unroll
    for (int e = 1; e < E_; ++e) if (q[e] > q[i0]) i0 = e;
    int i1 = (i0 == 0) ? 1 : 0;
#pragma unroll
    for (int e = 0; e < E_; ++e) if (e != i0 && q[e] > q[i1]) i1 = e;
    float v0 = q[i0], v1 = q[i1], den = v0 + v1 + 1e-8f;
    topi [token*K_+0] = i0;  topi [token*K_+1] = i1;
    gatev[token*K_+0] = v0/den; gatev[token*K_+1] = v1/den;
    atomicAdd(&counts[i0], 1);
    atomicAdd(&counts[i1], 1);
  }
}

__global__ void scan_kernel(const int* __restrict__ counts, int* __restrict__ offs,
                            int* __restrict__ fill){
  if (threadIdx.x == 0){
    int o = 0;
    for (int e = 0; e < E_; ++e){ offs[e] = o; o += counts[e]; fill[e] = 0; }
  }
}

__global__ void build_assign_kernel(const int* __restrict__ topi, const float* __restrict__ gatev,
                                    const int* __restrict__ offs, int* __restrict__ fill,
                                    int* __restrict__ atok, float* __restrict__ agate){
  int n = blockIdx.x*256 + threadIdx.x;
  if (n >= NTOK) return;
#pragma unroll
  for (int k = 0; k < K_; ++k){
    int e = topi[n*K_+k];
    int pos = offs[e] + atomicAdd(&fill[e], 1);
    atok[pos] = n;
    agate[pos] = gatev[n*K_+k];
  }
}

// ---------------- WMMA GEMM core (fragment-ready LDS + reg pipeline) ----------------
// Block tile 64x128, K-step 32; 8 waves as 2x4 grid of 32x32 wave tiles.
// A (16x32) mapping: (m,k) -> lane = (m&15) + ((k&16)?16:0), e = k&15
// B (32x16) mapping: (k,n) -> lane = ((k&16)?16:0) + (n&15), e = k&15
// K-loop is software pipelined: tile k+1 is fetched into registers while tile k computes.

#define GM_TM 64
#define GM_TN 128
#define GM_TK 32

__global__ __launch_bounds__(256) void gemm_bias_kernel(
    const float* __restrict__ A, const float* __restrict__ W,
    const float* __restrict__ bias, float* __restrict__ Cout,
    int M, int N, int Kd)
{
  __shared__ __align__(32) __bf16 Asf[4*FRAG_SZ];
  __shared__ __align__(32) __bf16 Bsf[8*FRAG_SZ];
  const int tid = threadIdx.x, lane = tid & 31, wv = tid >> 5;
  const int wm = wv >> 2, wn = wv & 3;
  const int row0 = blockIdx.y*GM_TM, col0 = blockIdx.x*GM_TN;

  // A staging assignment: r = row in tile, c = 8-wide k chunk
  const int ar_ = tid >> 2, ac_ = tid & 3;
  const bool a_ok = (row0 + ar_) < M;
  const float* a_row = &A[(size_t)(a_ok ? row0 + ar_ : 0)*Kd + ac_*8];
  __bf16* a_dst = &Asf[(ar_ >> 4)*FRAG_SZ + (((ar_ & 15) + ((ac_ & 2) ? 16 : 0))*16)
                       + ((ac_ & 1) ? 8 : 0)];
  // B staging assignment: n = column in tile, khalf selects 16-k run
  const int bn_ = tid & 127, bkh_ = tid >> 7;
  const int gc_ = col0 + bn_;
  const bool b_ok = gc_ < N;
  const float* b_col = &W[(size_t)(bkh_*16)*N + (b_ok ? gc_ : 0)];
  __bf16* b_dst = &Bsf[(bn_ >> 4)*FRAG_SZ + (bkh_*16 + (bn_ & 15))*16];

  v4f aA = {}, aB = {};
  float bR[16];
#pragma unroll
  for (int j = 0; j < 16; ++j) bR[j] = 0.f;

  if (a_ok){ aA = *(const v4f*)a_row; aB = *(const v4f*)(a_row + 4); }
  if (b_ok){
#pragma unroll
    for (int j = 0; j < 16; ++j) bR[j] = b_col[(size_t)j*N];
  }

  v8f acc[2][2] = {};
  for (int k0 = 0; k0 < Kd; k0 += GM_TK){
    { // commit registers (tile k0) to LDS in fragment layout
      v8bf ta;
#pragma unroll
      for (int j = 0; j < 4; ++j){ ta[j] = f2bf(aA[j]); ta[4+j] = f2bf(aB[j]); }
      *(v8bf*)a_dst = ta;
      v16bf tb;
#pragma unroll
      for (int j = 0; j < 16; ++j) tb[j] = f2bf(bR[j]);
      *(v16bf*)b_dst = tb;
    }
    __syncthreads();
    const int kn = k0 + GM_TK;
    if (kn < Kd){     // prefetch tile k0+32 into registers (overlaps WMMAs below)
      if (a_ok){ const float* ap = a_row + kn;
        aA = *(const v4f*)ap; aB = *(const v4f*)(ap + 4); }
      if (b_ok){ const float* wp = b_col + (size_t)kn*N;
#pragma unroll
        for (int j = 0; j < 16; ++j) bR[j] = wp[(size_t)j*N];
      }
    }
    v16bf a0 = load_frag(&Asf[(wm*2+0)*FRAG_SZ]);
    v16bf a1 = load_frag(&Asf[(wm*2+1)*FRAG_SZ]);
    v16bf b0 = load_frag(&Bsf[(wn*2+0)*FRAG_SZ]);
    v16bf b1 = load_frag(&Bsf[(wn*2+1)*FRAG_SZ]);
    acc[0][0] = wmma_bf16(a0, b0, acc[0][0]);
    acc[0][1] = wmma_bf16(a0, b1, acc[0][1]);
    acc[1][0] = wmma_bf16(a1, b0, acc[1][0]);
    acc[1][1] = wmma_bf16(a1, b1, acc[1][1]);
    __syncthreads();
  }
  const int nl = lane & 15, mb = (lane < 16) ? 0 : 8;
#pragma unroll
  for (int si = 0; si < 2; ++si)
#pragma unroll
    for (int sj = 0; sj < 2; ++sj)
#pragma unroll
      for (int r = 0; r < 8; ++r){
        int gm = row0 + wm*32 + si*16 + mb + r;
        int gn = col0 + wn*32 + sj*16 + nl;
        if (gm < M && gn < N)
          Cout[(size_t)gm*N + gn] = acc[si][sj][r] + (bias ? bias[gn] : 0.f);
      }
}

// Expert FC: gathered A rows (token list), gelu epilogue, writes h[assign_row][FF].
__global__ __launch_bounds__(256) void expert_fc_kernel(
    const float* __restrict__ xn, const float* __restrict__ wfc_l,
    const float* __restrict__ bfc_l, float* __restrict__ hbuf,
    const int* __restrict__ atok, const int* __restrict__ offs,
    const int* __restrict__ counts)
{
  const int e = blockIdx.z;
  const int cnt = counts[e];
  const int row0 = blockIdx.y*GM_TM;
  if (row0 >= cnt) return;
  const int base = offs[e];
  const float* W    = wfc_l + (size_t)e*C_*FF_;
  const float* bias = bfc_l + (size_t)e*FF_;
  const int col0 = blockIdx.x*GM_TN;

  __shared__ __align__(32) __bf16 Asf[4*FRAG_SZ];
  __shared__ __align__(32) __bf16 Bsf[8*FRAG_SZ];
  __shared__ int toks[GM_TM];
  const int tid = threadIdx.x, lane = tid & 31, wv = tid >> 5;
  const int wm = wv >> 2, wn = wv & 3;
  if (tid < GM_TM){
    int ar = row0 + tid;
    toks[tid] = (ar < cnt) ? atok[base + ar] : -1;
  }
  __syncthreads();

  const int ar_ = tid >> 2, ac_ = tid & 3;
  const int tok_ = toks[ar_];
  const bool a_ok = tok_ >= 0;
  const float* a_row = &xn[(size_t)(a_ok ? tok_ : 0)*C_ + ac_*8];
  __bf16* a_dst = &Asf[(ar_ >> 4)*FRAG_SZ + (((ar_ & 15) + ((ac_ & 2) ? 16 : 0))*16)
                       + ((ac_ & 1) ? 8 : 0)];
  const int bn_ = tid & 127, bkh_ = tid >> 7;
  const float* b_col = &W[(size_t)(bkh_*16)*FF_ + col0 + bn_];
  __bf16* b_dst = &Bsf[(bn_ >> 4)*FRAG_SZ + (bkh_*16 + (bn_ & 15))*16];

  v4f aA = {}, aB = {};
  float bR[16];
  if (a_ok){ aA = *(const v4f*)a_row; aB = *(const v4f*)(a_row + 4); }
#pragma unroll
  for (int j = 0; j < 16; ++j) bR[j] = b_col[(size_t)j*FF_];

  v8f acc[2][2] = {};
  for (int k0 = 0; k0 < C_; k0 += GM_TK){
    { v8bf ta;
#pragma unroll
      for (int j = 0; j < 4; ++j){ ta[j] = f2bf(aA[j]); ta[4+j] = f2bf(aB[j]); }
      *(v8bf*)a_dst = ta;
      v16bf tb;
#pragma unroll
      for (int j = 0; j < 16; ++j) tb[j] = f2bf(bR[j]);
      *(v16bf*)b_dst = tb;
    }
    __syncthreads();
    const int kn = k0 + GM_TK;
    if (kn < C_){
      if (a_ok){ const float* ap = a_row + kn;
        aA = *(const v4f*)ap; aB = *(const v4f*)(ap + 4); }
      const float* wp = b_col + (size_t)kn*FF_;
#pragma unroll
      for (int j = 0; j < 16; ++j) bR[j] = wp[(size_t)j*FF_];
    }
    v16bf a0 = load_frag(&Asf[(wm*2+0)*FRAG_SZ]);
    v16bf a1 = load_frag(&Asf[(wm*2+1)*FRAG_SZ]);
    v16bf b0 = load_frag(&Bsf[(wn*2+0)*FRAG_SZ]);
    v16bf b1 = load_frag(&Bsf[(wn*2+1)*FRAG_SZ]);
    acc[0][0] = wmma_bf16(a0, b0, acc[0][0]);
    acc[0][1] = wmma_bf16(a0, b1, acc[0][1]);
    acc[1][0] = wmma_bf16(a1, b0, acc[1][0]);
    acc[1][1] = wmma_bf16(a1, b1, acc[1][1]);
    __syncthreads();
  }
  const int nl = lane & 15, mb = (lane < 16) ? 0 : 8;
#pragma unroll
  for (int si = 0; si < 2; ++si)
#pragma unroll
    for (int sj = 0; sj < 2; ++sj)
#pragma unroll
      for (int r = 0; r < 8; ++r){
        int lm = wm*32 + si*16 + mb + r;
        int ar = row0 + lm;
        int gn = col0 + wn*32 + sj*16 + nl;
        if (ar < cnt)
          hbuf[(size_t)(base+ar)*FF_ + gn] = gelu_new(acc[si][sj][r] + bias[gn]);
      }
}

// Expert PROJ: A = contiguous h rows; epilogue scatters gate*(acc+bias) into x via atomicAdd.
__global__ __launch_bounds__(256) void expert_proj_kernel(
    const float* __restrict__ hbuf, const float* __restrict__ wproj_l,
    const float* __restrict__ bproj_l, float* __restrict__ xres,
    const int* __restrict__ atok, const float* __restrict__ agate,
    const int* __restrict__ offs, const int* __restrict__ counts)
{
  const int e = blockIdx.z;
  const int cnt = counts[e];
  const int row0 = blockIdx.y*GM_TM;
  if (row0 >= cnt) return;
  const int base = offs[e];
  const float* W    = wproj_l + (size_t)e*FF_*C_;
  const float* bias = bproj_l + (size_t)e*C_;
  const int col0 = blockIdx.x*GM_TN;

  __shared__ __align__(32) __bf16 Asf[4*FRAG_SZ];
  __shared__ __align__(32) __bf16 Bsf[8*FRAG_SZ];
  const int tid = threadIdx.x, lane = tid & 31, wv = tid >> 5;
  const int wm = wv >> 2, wn = wv & 3;

  const int ar_ = tid >> 2, ac_ = tid & 3;
  const bool a_ok = (row0 + ar_) < cnt;
  const float* a_row = &hbuf[(size_t)(base + (a_ok ? row0 + ar_ : 0))*FF_ + ac_*8];
  __bf16* a_dst = &Asf[(ar_ >> 4)*FRAG_SZ + (((ar_ & 15) + ((ac_ & 2) ? 16 : 0))*16)
                       + ((ac_ & 1) ? 8 : 0)];
  const int bn_ = tid & 127, bkh_ = tid >> 7;
  const float* b_col = &W[(size_t)(bkh_*16)*C_ + col0 + bn_];   // N = C_, always valid
  __bf16* b_dst = &Bsf[(bn_ >> 4)*FRAG_SZ + (bkh_*16 + (bn_ & 15))*16];

  v4f aA = {}, aB = {};
  float bR[16];
  if (a_ok){ aA = *(const v4f*)a_row; aB = *(const v4f*)(a_row + 4); }
#pragma unroll
  for (int j = 0; j < 16; ++j) bR[j] = b_col[(size_t)j*C_];

  v8f acc[2][2] = {};
  for (int k0 = 0; k0 < FF_; k0 += GM_TK){
    { v8bf ta;
#pragma unroll
      for (int j = 0; j < 4; ++j){ ta[j] = f2bf(aA[j]); ta[4+j] = f2bf(aB[j]); }
      *(v8bf*)a_dst = ta;
      v16bf tb;
#pragma unroll
      for (int j = 0; j < 16; ++j) tb[j] = f2bf(bR[j]);
      *(v16bf*)b_dst = tb;
    }
    __syncthreads();
    const int kn = k0 + GM_TK;
    if (kn < FF_){
      if (a_ok){ const float* ap = a_row + kn;
        aA = *(const v4f*)ap; aB = *(const v4f*)(ap + 4); }
      const float* wp = b_col + (size_t)kn*C_;
#pragma unroll
      for (int j = 0; j < 16; ++j) bR[j] = wp[(size_t)j*C_];
    }
    v16bf a0 = load_frag(&Asf[(wm*2+0)*FRAG_SZ]);
    v16bf a1 = load_frag(&Asf[(wm*2+1)*FRAG_SZ]);
    v16bf b0 = load_frag(&Bsf[(wn*2+0)*FRAG_SZ]);
    v16bf b1 = load_frag(&Bsf[(wn*2+1)*FRAG_SZ]);
    acc[0][0] = wmma_bf16(a0, b0, acc[0][0]);
    acc[0][1] = wmma_bf16(a0, b1, acc[0][1]);
    acc[1][0] = wmma_bf16(a1, b0, acc[1][0]);
    acc[1][1] = wmma_bf16(a1, b1, acc[1][1]);
    __syncthreads();
  }
  const int nl = lane & 15, mb = (lane < 16) ? 0 : 8;
#pragma unroll
  for (int si = 0; si < 2; ++si)
#pragma unroll
    for (int sj = 0; sj < 2; ++sj)
#pragma unroll
      for (int r = 0; r < 8; ++r){
        int lm = wm*32 + si*16 + mb + r;
        int ar = row0 + lm;
        int gn = col0 + wn*32 + sj*16 + nl;
        if (ar < cnt){
          int tok = atok[base+ar];
          float g = agate[base+ar];
          atomicAdd(&xres[(size_t)tok*C_ + gn], g*(acc[si][sj][r] + bias[gn]));
        }
      }
}

// ---------------- flash attention ----------------
// Block = (64 queries) x (HD=64) for one (b,h); streams 64-key tiles with online softmax.
// All matrix tiles kept in fragment-ready LDS layout.
#define AS_LD 65

__global__ __launch_bounds__(256) void attn_kernel(const float* __restrict__ qkv,
                                                   float* __restrict__ out)
{
  const int qt = blockIdx.x;      // query tile (0..15)
  const int h  = blockIdx.y;
  const int b  = blockIdx.z;
  const int tid = threadIdx.x, lane = tid & 31, wv = tid >> 5;
  const int wm = wv >> 2, wn = wv & 3;   // waves: 2x4 -> 32x16 output tiles

  __shared__ __align__(32) __bf16 Qsf[8*FRAG_SZ];  // [msub(4)][dblk(2)]
  __shared__ __align__(32) __bf16 KTf[8*FRAG_SZ];  // [key-nsub(4)][dblk(2)]
  __shared__ __align__(32) __bf16 Vsf[8*FRAG_SZ];  // [d-nsub(4)][keyblk(2)]
  __shared__ __align__(32) __bf16 Psf[8*FRAG_SZ];  // [msub(4)][keyblk(2)]
  __shared__ float  Sls[64*AS_LD];
  __shared__ float  mrow[64], lrow[64], arow[64], red[64*4];

  const int qbase = qt*64;
  const size_t rs = 3*C_;                         // qkv row stride
  const float* qp = qkv + (size_t)b*T_*rs;        // batch base

  // stage Q (A-layout; scale 1/sqrt(64) folded in): 512 chunks of 8, 2 per thread
#pragma unroll
  for (int cc = tid; cc < 512; cc += 256){
    int r = cc >> 3, c = cc & 7;
    int kblk = c >> 2, c2 = c & 3;
    __bf16* dst = &Qsf[((r >> 4)*2 + kblk)*FRAG_SZ
                       + (((r & 15) + ((c2 & 2) ? 16 : 0))*16) + ((c2 & 1) ? 8 : 0)];
    const float* ap = &qp[(size_t)(qbase+r)*rs + h*HD_ + c*8];
    v4f q0 = *(const v4f*)ap, q1 = *(const v4f*)(ap + 4);
    v8bf tv;
#pragma unroll
    for (int j = 0; j < 4; ++j){ tv[j] = f2bf(q0[j]*0.125f); tv[4+j] = f2bf(q1[j]*0.125f); }
    *(v8bf*)dst = tv;
  }
  if (tid < 64){ mrow[tid] = -3.0e38f; lrow[tid] = 0.f; }
  v8f oc0 = {}, oc1 = {};
  __syncthreads();

  for (int kb = 0; kb <= qt; ++kb){
    const int kbase = kb*64;
    { // stage K^T (B-layout: n=key, k=d): thread = (key, 16-d segment), 64B contiguous
      int n = tid & 63, dseg = tid >> 6;
      int kblk = dseg >> 1, khalf = dseg & 1;
      __bf16* dst = &KTf[((n >> 4)*2 + kblk)*FRAG_SZ + (khalf*16 + (n & 15))*16];
      const float* ap = &qp[(size_t)(kbase+n)*rs + C_ + h*HD_ + dseg*16];
      v4f k0v = *(const v4f*)ap,     k1v = *(const v4f*)(ap + 4);
      v4f k2v = *(const v4f*)(ap+8), k3v = *(const v4f*)(ap + 12);
      v16bf tv;
#pragma unroll
      for (int j = 0; j < 4; ++j){
        tv[j] = f2bf(k0v[j]); tv[4+j] = f2bf(k1v[j]);
        tv[8+j] = f2bf(k2v[j]); tv[12+j] = f2bf(k3v[j]);
      }
      *(v16bf*)dst = tv;
    }
    { // stage V (B-layout: n=d, k=key): thread = (d, 16-key segment), coalesced across lanes
      int n = tid & 63, kseg = tid >> 6;
      int kblk = kseg >> 1, khalf = kseg & 1;
      __bf16* dst = &Vsf[((n >> 4)*2 + kblk)*FRAG_SZ + (khalf*16 + (n & 15))*16];
      const float* ap = &qp[(size_t)(kbase + kseg*16)*rs + 2*C_ + h*HD_ + n];
      v16bf tv;
#pragma unroll
      for (int j = 0; j < 16; ++j) tv[j] = f2bf(ap[(size_t)j*rs]);
      *(v16bf*)dst = tv;
    }
    __syncthreads();

    // S = Q K^T (wave: query rows wm*32..+32, key cols wn*16..+16)
    v8f s0 = {}, s1 = {};
#pragma unroll
    for (int kblk = 0; kblk < 2; ++kblk){
      v16bf a0 = load_frag(&Qsf[((wm*2+0)*2 + kblk)*FRAG_SZ]);
      v16bf a1 = load_frag(&Qsf[((wm*2+1)*2 + kblk)*FRAG_SZ]);
      v16bf bb = load_frag(&KTf[(wn*2 + kblk)*FRAG_SZ]);
      s0 = wmma_bf16(a0, bb, s0);
      s1 = wmma_bf16(a1, bb, s1);
    }
    { const int n = wn*16 + (lane & 15), mb = (lane < 16) ? 0 : 8;
#pragma unroll
      for (int r = 0; r < 8; ++r){
        Sls[(wm*32    + mb + r)*AS_LD + n] = s0[r];
        Sls[(wm*32+16 + mb + r)*AS_LD + n] = s1[r];
      }
    }
    __syncthreads();

    if (kb == qt){                       // causal mask on diagonal tile
      for (int i = tid; i < 64*64; i += 256){
        int r = i >> 6, c = i & 63;
        if (c > r) Sls[r*AS_LD + c] = -3.0e38f;
      }
      __syncthreads();
    }

    // online softmax: 4 threads per row, 16 cols each
    { int r = tid >> 2, seg = tid & 3;
      float mx = -3.0e38f;
#pragma unroll
      for (int j = 0; j < 16; ++j) mx = fmaxf(mx, Sls[r*AS_LD + seg*16 + j]);
      red[r*4 + seg] = mx;
    }
    __syncthreads();
    if ((tid & 3) == 0){
      int r = tid >> 2;
      float mx = fmaxf(fmaxf(red[r*4], red[r*4+1]), fmaxf(red[r*4+2], red[r*4+3]));
      float nm = fmaxf(mrow[r], mx);
      arow[r] = __expf(mrow[r] - nm);
      mrow[r] = nm;
    }
    __syncthreads();
    { // exp + write P in fragment layout: two contiguous 8-elem runs per thread
      int r = tid >> 2, seg = tid & 3;
      float nm = mrow[r], s = 0.f;
      int kblk = seg >> 1;
      int eoff = (seg & 1) ? 8 : 0;
      __bf16* fb = &Psf[((r >> 4)*2 + kblk)*FRAG_SZ];
      v8bf lo8, hi8;
#pragma unroll
      for (int j = 0; j < 8; ++j){
        float p = __expf(Sls[r*AS_LD + seg*16 + j] - nm);
        lo8[j] = f2bf(p); s += p;
      }
#pragma unroll
      for (int j = 8; j < 16; ++j){
        float p = __expf(Sls[r*AS_LD + seg*16 + j] - nm);
        hi8[j-8] = f2bf(p); s += p;
      }
      *(v8bf*)(fb + ((r & 15) + 2*eoff)*16    ) = lo8;   // elements 0-7 of lane half (seg&1)
      *(v8bf*)(fb + ((r & 15) + 2*eoff)*16 + 8) = hi8;   // elements 8-15 of lane half (seg&1)
      red[r*4 + seg] = s;
    }
    __syncthreads();
    if ((tid & 3) == 0){
      int r = tid >> 2;
      lrow[r] = lrow[r]*arow[r] + red[r*4] + red[r*4+1] + red[r*4+2] + red[r*4+3];
    }
    __syncthreads();

    // rescale O accumulators by alpha(row)
    { const int mb = (lane < 16) ? 0 : 8;
#pragma unroll
      for (int r = 0; r < 8; ++r){
        oc0[r] *= arow[wm*32      + mb + r];
        oc1[r] *= arow[wm*32 + 16 + mb + r];
      }
    }
    // O += P @ V (wave: query rows wm*32..+32, d cols wn*16..+16)
#pragma unroll
    for (int kblk = 0; kblk < 2; ++kblk){
      v16bf a0 = load_frag(&Psf[((wm*2+0)*2 + kblk)*FRAG_SZ]);
      v16bf a1 = load_frag(&Psf[((wm*2+1)*2 + kblk)*FRAG_SZ]);
      v16bf bb = load_frag(&Vsf[(wn*2 + kblk)*FRAG_SZ]);
      oc0 = wmma_bf16(a0, bb, oc0);
      oc1 = wmma_bf16(a1, bb, oc1);
    }
    __syncthreads();
  }

  { const int n = wn*16 + (lane & 15), mb = (lane < 16) ? 0 : 8;
#pragma unroll
    for (int r = 0; r < 8; ++r){
      int m0 = wm*32      + mb + r;
      int m1 = wm*32 + 16 + mb + r;
      out[(size_t)(b*T_ + qbase + m0)*C_ + h*HD_ + n] = oc0[r] / lrow[m0];
      out[(size_t)(b*T_ + qbase + m1)*C_ + h*HD_ + n] = oc1[r] / lrow[m1];
    }
  }
}

// ---------------- host orchestration ----------------

extern "C" void kernel_launch(void* const* d_in, const int* in_sizes, int n_in,
                              void* d_out, int out_size, void* d_ws, size_t ws_size,
                              hipStream_t stream)
{
  (void)in_sizes; (void)n_in; (void)out_size; (void)ws_size;
  const int*   idx     = (const int*)  d_in[0];
  const float* wte     = (const float*)d_in[1];
  const float* wpe     = (const float*)d_in[2];
  const float* ln1_g   = (const float*)d_in[3];
  const float* ln1_b   = (const float*)d_in[4];
  const float* w_qkv   = (const float*)d_in[5];
  const float* b_qkv   = (const float*)d_in[6];
  const float* w_o     = (const float*)d_in[7];
  const float* b_o     = (const float*)d_in[8];
  const float* ln2_g   = (const float*)d_in[9];
  const float* ln2_b   = (const float*)d_in[10];
  const float* w_router= (const float*)d_in[11];
  const float* w_fc    = (const float*)d_in[12];
  const float* b_fc    = (const float*)d_in[13];
  const float* w_proj  = (const float*)d_in[14];
  const float* b_proj  = (const float*)d_in[15];
  const float* lnf_g   = (const float*)d_in[16];
  const float* lnf_b   = (const float*)d_in[17];
  const float* w_head  = (const float*)d_in[18];
  float* logits = (float*)d_out;

  char* wsp = (char*)d_ws;
  auto alloc = [&](size_t bytes)->void*{
    void* p = (void*)wsp; wsp += ((bytes + 255) & ~(size_t)255); return p;
  };
  float* x      = (float*)alloc((size_t)NTOK*C_*4);
  float* xn     = (float*)alloc((size_t)NTOK*C_*4);
  float* qkv    = (float*)alloc((size_t)NTOK*3*C_*4);
  float* attnout= (float*)alloc((size_t)NTOK*C_*4);
  float* ybuf   = (float*)alloc((size_t)NTOK*C_*4);
  float* hbuf   = (float*)alloc((size_t)NASSIGN*FF_*4);
  int*   topi   = (int*)  alloc((size_t)NTOK*K_*4);
  float* gatev  = (float*)alloc((size_t)NTOK*K_*4);
  int*   counts = (int*)  alloc(E_*4);
  int*   offs   = (int*)  alloc(E_*4);
  int*   fill   = (int*)  alloc(E_*4);
  int*   atok   = (int*)  alloc((size_t)NASSIGN*4);
  float* agate  = (float*)alloc((size_t)NASSIGN*4);

  embed_kernel<<<(NTOK*C_)/256, 256, 0, stream>>>(idx, wte, wpe, x);

  for (int l = 0; l < L_; ++l){
    ln_kernel<<<NTOK, 256, 0, stream>>>(x, ln1_g + (size_t)l*C_, ln1_b + (size_t)l*C_, xn);

    gemm_bias_kernel<<<dim3((3*C_)/GM_TN, NTOK/GM_TM), 256, 0, stream>>>(
        xn, w_qkv + (size_t)l*C_*3*C_, b_qkv + (size_t)l*3*C_, qkv, NTOK, 3*C_, C_);

    attn_kernel<<<dim3(T_/64, NH_, B_), 256, 0, stream>>>(qkv, attnout);

    gemm_bias_kernel<<<dim3(C_/GM_TN, NTOK/GM_TM), 256, 0, stream>>>(
        attnout, w_o + (size_t)l*C_*C_, b_o + (size_t)l*C_, ybuf, NTOK, C_, C_);

    add_kernel<<<(NTOK*C_)/256, 256, 0, stream>>>(x, ybuf, NTOK*C_);

    ln_kernel<<<NTOK, 256, 0, stream>>>(x, ln2_g + (size_t)l*C_, ln2_b + (size_t)l*C_, xn);

    zero_counts_kernel<<<1, 32, 0, stream>>>(counts);
    router_kernel<<<NTOK/8, 256, 0, stream>>>(xn, w_router + (size_t)l*C_*E_,
                                              topi, gatev, counts);
    scan_kernel<<<1, 32, 0, stream>>>(counts, offs, fill);
    build_assign_kernel<<<NTOK/256, 256, 0, stream>>>(topi, gatev, offs, fill, atok, agate);

    expert_fc_kernel<<<dim3(FF_/GM_TN, NASSIGN/GM_TM, E_), 256, 0, stream>>>(
        xn, w_fc + (size_t)l*E_*C_*FF_, b_fc + (size_t)l*E_*FF_,
        hbuf, atok, offs, counts);

    expert_proj_kernel<<<dim3(C_/GM_TN, NASSIGN/GM_TM, E_), 256, 0, stream>>>(
        hbuf, w_proj + (size_t)l*E_*FF_*C_, b_proj + (size_t)l*E_*C_,
        x, atok, agate, offs, counts);
  }

  ln_kernel<<<NTOK, 256, 0, stream>>>(x, lnf_g, lnf_b, xn);
  gemm_bias_kernel<<<dim3((V_ + GM_TN - 1)/GM_TN, NTOK/GM_TM), 256, 0, stream>>>(
      xn, w_head, nullptr, logits, NTOK, V_, C_);
}
